// MyAttentionHead_16492674417444
// MI455X (gfx1250) — matrix operand order repaired and checked by reading, not compiled, not run on hardware
//
#include <hip/hip_runtime.h>
#include <hip/hip_bf16.h>

#define D_MODEL 1024
#define D_HEAD  64
#define BATCH   8
#define SEQ     2048

typedef __attribute__((ext_vector_type(16))) _Float16 v16h;
typedef __attribute__((ext_vector_type(8)))  _Float16 v8h;
typedef __attribute__((ext_vector_type(8)))  float    v8f;

union HFrag { v16h v; v8h h[2]; };

static __device__ __forceinline__ v8f wmma_f16(v16h a, v16h b, v8f c) {
  // D = A(16x32 f16) * B(32x16 f16) + C(16x16 f32)
  return __builtin_amdgcn_wmma_f32_16x16x32_f16(false, a, false, b, (short)0, c,
                                                false, false);
}

// ---------------------------------------------------------------------------
// Kernel 1: Out[m,n] (f16) = sum_k X[m,k] * W[n,k]   (M=B*S, N=64, K=1024)
// Block = 256 threads (8 waves); wave w owns rows [blk*128 + 16w, +16), all 64 cols.
// ---------------------------------------------------------------------------
__global__ void __launch_bounds__(256)
proj_kernel(const float* __restrict__ X, const float* __restrict__ W,
            _Float16* __restrict__ Out) {
  const int lane = threadIdx.x & 31;
  const int wave = threadIdx.x >> 5;
  const int m0   = blockIdx.x * 128 + wave * 16;
  const int lr   = lane & 15;
  const int hi   = lane >> 4;

  v8f acc[4] = {};
  for (int k0 = 0; k0 < D_MODEL; k0 += 32) {
    HFrag a;
    {
      const float* p = X + (size_t)(m0 + lr) * D_MODEL + k0 + hi * 8;
#pragma unroll
      for (int j = 0; j < 8; ++j) a.h[0][j] = (_Float16)p[j];
#pragma unroll
      for (int j = 0; j < 8; ++j) a.h[1][j] = (_Float16)p[16 + j];
    }
#pragma unroll
    for (int ct = 0; ct < 4; ++ct) {
      HFrag bf;
      const float* p = W + (size_t)(ct * 16 + lr) * D_MODEL + k0 + hi * 8;
#pragma unroll
      for (int j = 0; j < 8; ++j) bf.h[0][j] = (_Float16)p[j];
#pragma unroll
      for (int j = 0; j < 8; ++j) bf.h[1][j] = (_Float16)p[16 + j];
      acc[ct] = wmma_f16(a.v, bf.v, acc[ct]);
    }
  }
#pragma unroll
  for (int ct = 0; ct < 4; ++ct)
#pragma unroll
    for (int r = 0; r < 8; ++r) {
      const int row = m0 + r + 8 * hi;
      const int col = ct * 16 + lr;
      Out[(size_t)row * D_HEAD + col] = (_Float16)acc[ct][r];
    }
}

// ---------------------------------------------------------------------------
// Kernel 2: XvT[b,d,s] (f16) = sum_h Wo[d,h] * V[b*S+s,h]  (stored transposed!)
// ---------------------------------------------------------------------------
__global__ void __launch_bounds__(256)
xvt_kernel(const float* __restrict__ Wo, const _Float16* __restrict__ V,
           _Float16* __restrict__ XvT) {
  const int lane = threadIdx.x & 31;
  const int wave = threadIdx.x >> 5;
  const int b    = blockIdx.z;
  const int d0   = blockIdx.y * 16;
  const int s0   = blockIdx.x * 128 + wave * 16;
  const int lr   = lane & 15;
  const int hi   = lane >> 4;

  v8f acc = {};
#pragma unroll
  for (int k0 = 0; k0 < D_HEAD; k0 += 32) {
    HFrag a;
    const float* pa = Wo + (size_t)(d0 + lr) * D_HEAD + k0 + hi * 8;
#pragma unroll
    for (int j = 0; j < 8; ++j) a.h[0][j] = (_Float16)pa[j];
#pragma unroll
    for (int j = 0; j < 8; ++j) a.h[1][j] = (_Float16)pa[16 + j];
    HFrag bf;
    const _Float16* pb = V + (size_t)(b * SEQ + s0 + lr) * D_HEAD + k0 + hi * 8;
    bf.h[0] = *(const v8h*)(pb);
    bf.h[1] = *(const v8h*)(pb + 16);
    acc = wmma_f16(a.v, bf.v, acc);
  }
#pragma unroll
  for (int r = 0; r < 8; ++r) {
    const int d = d0 + r + 8 * hi;
    XvT[((size_t)b * D_MODEL + d) * SEQ + s0 + lr] = (_Float16)acc[r];
  }
}

// ---------------------------------------------------------------------------
// Kernel 3: P[b,q,k] = exp(scale * q.k) for k<=q else 0 (f16, unnormalized);
//           linv[b,q] = 1 / sum_k P. Block owns 16 q-rows; wave w takes
//           k-tiles w, w+8, ... over the full row (zero-fills masked tiles).
// ---------------------------------------------------------------------------
__global__ void __launch_bounds__(256)
scores_kernel(const _Float16* __restrict__ Q, const _Float16* __restrict__ K,
              _Float16* __restrict__ P, float* __restrict__ linv) {
  __shared__ float lred[8 * 16];
  const int lane = threadIdx.x & 31;
  const int wave = threadIdx.x >> 5;
  const int b    = blockIdx.y;
  const int q0   = blockIdx.x * 16;
  const int lr   = lane & 15;
  const int hi   = lane >> 4;
  const float scale = 0.03125f; // 1/sqrt(1024)

  // Q fragments for rows q0..q0+15, head dims 0..63 (two K=32 A-frags)
  HFrag a0, a1;
  const _Float16* pq = Q + (size_t)(b * SEQ + q0 + lr) * D_HEAD + hi * 8;
  a0.h[0] = *(const v8h*)(pq);
  a0.h[1] = *(const v8h*)(pq + 16);
  a1.h[0] = *(const v8h*)(pq + 32);
  a1.h[1] = *(const v8h*)(pq + 48);

  float psum[8] = {0.f, 0.f, 0.f, 0.f, 0.f, 0.f, 0.f, 0.f};

  for (int kt = wave; kt < SEQ / 16; kt += 8) {
    const int k0 = kt * 16;
    if (k0 > q0 + 15) { // fully masked tile: define P as zero
#pragma unroll
      for (int r = 0; r < 8; ++r) {
        const int q = q0 + r + 8 * hi;
        P[((size_t)(b * SEQ + q)) * SEQ + k0 + lr] = (_Float16)0.f;
      }
      continue;
    }
    HFrag b0, b1;
    const _Float16* pk = K + (size_t)(b * SEQ + k0 + lr) * D_HEAD + hi * 8;
    b0.h[0] = *(const v8h*)(pk);
    b0.h[1] = *(const v8h*)(pk + 16);
    b1.h[0] = *(const v8h*)(pk + 32);
    b1.h[1] = *(const v8h*)(pk + 48);
    v8f c = {};
    c = wmma_f16(a0.v, b0.v, c);
    c = wmma_f16(a1.v, b1.v, c);
#pragma unroll
    for (int r = 0; r < 8; ++r) {
      const int q = q0 + r + 8 * hi;
      const int k = k0 + lr;
      const float p = (k <= q) ? __expf(c[r] * scale) : 0.f;
      psum[r] += p;
      P[((size_t)(b * SEQ + q)) * SEQ + k] = (_Float16)p;
    }
  }
  // reduce row sums across the 16 lanes of each half-wave
#pragma unroll
  for (int m = 1; m < 16; m <<= 1)
#pragma unroll
    for (int r = 0; r < 8; ++r) psum[r] += __shfl_xor(psum[r], m, 32);
  if (lr == 0) {
#pragma unroll
    for (int r = 0; r < 8; ++r) lred[wave * 16 + r + 8 * hi] = psum[r];
  }
  __syncthreads();
  if (threadIdx.x < 16) {
    float s = 0.f;
#pragma unroll
    for (int w = 0; w < 8; ++w) s += lred[w * 16 + (int)threadIdx.x];
    linv[b * SEQ + q0 + threadIdx.x] = 1.0f / s;
  }
}

// ---------------------------------------------------------------------------
// Kernel 4: out[b,q,d] = (sum_k P[b,q,k] * XvT[b,d,k]) * linv[b,q]
// 128x128 output block, 8 waves in a 4x2 grid (each wave: 32 rows x 64 cols,
// 8 accumulator tiles). Causal: k-loop stops at the block diagonal.
// ---------------------------------------------------------------------------
__global__ void __launch_bounds__(256)
out_kernel(const _Float16* __restrict__ P, const _Float16* __restrict__ XvT,
           const float* __restrict__ linv, float* __restrict__ out) {
  const int lane = threadIdx.x & 31;
  const int wave = threadIdx.x >> 5;
  const int b    = blockIdx.z;
  const int Q0   = blockIdx.x * 128;
  const int N0   = blockIdx.y * 128;
  const int R0   = Q0 + (wave >> 1) * 32;
  const int C0   = N0 + (wave & 1) * 64;
  const int lr   = lane & 15;
  const int hi   = lane >> 4;

  const _Float16* Pb = P + (size_t)b * SEQ * SEQ;
  const _Float16* Xb = XvT + (size_t)b * D_MODEL * SEQ;

  v8f acc[2][4] = {};
  const int kend = Q0 + 128; // P is zero for k > q, undefined-free up to kend
  for (int k0 = 0; k0 < kend; k0 += 32) {
    HFrag a[2];
#pragma unroll
    for (int rt = 0; rt < 2; ++rt) {
      const _Float16* p = Pb + (size_t)(R0 + rt * 16 + lr) * SEQ + k0 + hi * 8;
      a[rt].h[0] = *(const v8h*)(p);
      a[rt].h[1] = *(const v8h*)(p + 16);
    }
    HFrag bf[4];
#pragma unroll
    for (int ct = 0; ct < 4; ++ct) {
      const _Float16* p = Xb + (size_t)(C0 + ct * 16 + lr) * SEQ + k0 + hi * 8;
      bf[ct].h[0] = *(const v8h*)(p);
      bf[ct].h[1] = *(const v8h*)(p + 16);
    }
#pragma unroll
    for (int rt = 0; rt < 2; ++rt)
#pragma unroll
      for (int ct = 0; ct < 4; ++ct)
        acc[rt][ct] = wmma_f16(a[rt].v, bf[ct].v, acc[rt][ct]);
  }
#pragma unroll
  for (int rt = 0; rt < 2; ++rt)
#pragma unroll
    for (int r = 0; r < 8; ++r) {
      const int q = R0 + rt * 16 + r + 8 * hi;
      const float li = linv[b * SEQ + q];
      float* po = out + ((size_t)(b * SEQ + q)) * D_MODEL + C0 + lr;
#pragma unroll
      for (int ct = 0; ct < 4; ++ct) po[ct * 16] = acc[rt][ct][r] * li;
    }
}

// ---------------------------------------------------------------------------
extern "C" void kernel_launch(void* const* d_in, const int* in_sizes, int n_in,
                              void* d_out, int out_size, void* d_ws, size_t ws_size,
                              hipStream_t stream) {
  const float* X  = (const float*)d_in[0];
  const float* Wq = (const float*)d_in[1];
  const float* Wk = (const float*)d_in[2];
  const float* Wv = (const float*)d_in[3];
  const float* Wo = (const float*)d_in[4];
  float* out = (float*)d_out;

  char* ws = (char*)d_ws;
  size_t off = 0;
  auto alloc = [&](size_t bytes) -> void* {
    void* p = ws + off;
    off = (off + bytes + 255) & ~(size_t)255;
    return p;
  };
  _Float16* Qb   = (_Float16*)alloc((size_t)BATCH * SEQ * D_HEAD * 2);
  _Float16* Kb   = (_Float16*)alloc((size_t)BATCH * SEQ * D_HEAD * 2);
  _Float16* Vb   = (_Float16*)alloc((size_t)BATCH * SEQ * D_HEAD * 2);
  _Float16* XvT  = (_Float16*)alloc((size_t)BATCH * D_MODEL * SEQ * 2);
  _Float16* Pbuf = (_Float16*)alloc((size_t)BATCH * SEQ * SEQ * 2);
  float*    linv = (float*)alloc((size_t)BATCH * SEQ * 4);

  dim3 blk(256);
  proj_kernel<<<(BATCH * SEQ) / 128, blk, 0, stream>>>(X, Wq, Qb);
  proj_kernel<<<(BATCH * SEQ) / 128, blk, 0, stream>>>(X, Wk, Kb);
  proj_kernel<<<(BATCH * SEQ) / 128, blk, 0, stream>>>(X, Wv, Vb);
  xvt_kernel<<<dim3(SEQ / 128, D_MODEL / 16, BATCH), blk, 0, stream>>>(Wo, Vb, XvT);
  scores_kernel<<<dim3(SEQ / 16, BATCH), blk, 0, stream>>>(Qb, Kb, Pbuf, linv);
  out_kernel<<<dim3(SEQ / 128, D_MODEL / 128, BATCH), blk, 0, stream>>>(Pbuf, XvT, linv, out);
}
